// Spatial_MGCN_27315992002880
// MI455X (gfx1250) — compile-verified, matrix operand order checked by reading, not run on hardware
//
#include <hip/hip_runtime.h>
#include <hip/hip_bf16.h>
#include <math.h>

typedef __bf16 bf16;
typedef __attribute__((ext_vector_type(16))) __bf16 v16bf;
typedef __attribute__((ext_vector_type(8)))  __bf16 v8bf;
typedef __attribute__((ext_vector_type(8)))  float  v8f;

#define TM 64
#define TN 64
#define TK 32

// ---------------------------------------------------------------------------
// Conversions
// ---------------------------------------------------------------------------
// out[row][k<Kp] = bf16(in[row][k<K]) with zero pad for k in [K,Kp)
__global__ void cvt_pad_bf16(const float* __restrict__ in, bf16* __restrict__ out,
                             int K, int Kp) {
    int k = blockIdx.x * blockDim.x + threadIdx.x;
    long row = blockIdx.y;
    if (k >= Kp) return;
    out[row * Kp + k] = (k < K) ? (bf16)in[row * (long)K + k] : (bf16)0.0f;
}

// Wt[n][k<Kp] = bf16(W[k<K][n]) with zero pad  (W is [K][Nc] row-major)
__global__ void cvt_transpose_bf16(const float* __restrict__ W, bf16* __restrict__ Wt,
                                   int K, int Nc, int Kp) {
    int k = blockIdx.x * blockDim.x + threadIdx.x;
    int n = blockIdx.y;
    if (k >= Kp) return;
    Wt[(long)n * Kp + k] = (k < K) ? (bf16)W[(long)k * Nc + n] : (bf16)0.0f;
}

// ---------------------------------------------------------------------------
// Elementwise helpers
// ---------------------------------------------------------------------------
__global__ void bias_relu_cvt(const float* __restrict__ acc, const float* __restrict__ bias,
                              bf16* __restrict__ out, long n, int hmask) {
    long i = (long)blockIdx.x * blockDim.x + threadIdx.x;
    if (i >= n) return;
    float v = acc[i] + bias[i & hmask];
    out[i] = (bf16)fmaxf(v, 0.0f);
}

__global__ void bias_store(const float* __restrict__ acc, const float* __restrict__ bias,
                           float* __restrict__ out, long n, int hmask) {
    long i = (long)blockIdx.x * blockDim.x + threadIdx.x;
    if (i >= n) return;
    out[i] = acc[i] + bias[i & hmask];
}

// ---------------------------------------------------------------------------
// SpMM: out[dst] += w * h[src].  One block per edge, blockDim == H: edge
// metadata is block-uniform (scalar loads) and h-read / atomic are coalesced.
// ---------------------------------------------------------------------------
__global__ void spmm_edge(const int* __restrict__ src, const int* __restrict__ dst,
                          const float* __restrict__ w, const float* __restrict__ h,
                          float* __restrict__ out, int H) {
    int e = blockIdx.x;
    int c = threadIdx.x;
    int s = src[e];
    int d = dst[e];
    float wv = w[e];
    unsafeAtomicAdd(&out[(long)d * H + c], wv * h[(long)s * H + c]);
}

// ---------------------------------------------------------------------------
// Attention fusion: beta = softmax(tanh(z@Wa1+ba1)@Wa2); emb = sum beta*z
// ---------------------------------------------------------------------------
__global__ void attention_fuse(const float* __restrict__ emb1, const float* __restrict__ emb2,
                               const float* __restrict__ Wa1, const float* __restrict__ ba1,
                               const float* __restrict__ Wa2, bf16* __restrict__ outb, int n) {
    int i = blockIdx.x * blockDim.x + threadIdx.x;
    if (i >= n) return;
    const float* e0 = emb1 + (long)i * 64;
    const float* e1 = emb2 + (long)i * 64;
    float logit[2];
    #pragma unroll
    for (int t = 0; t < 2; ++t) {
        const float* e = t == 0 ? e0 : e1;
        float acc = 0.0f;
        for (int j = 0; j < 16; ++j) {
            float s = ba1[j];
            for (int k = 0; k < 64; ++k) s += e[k] * Wa1[k * 16 + j];
            acc += tanhf(s) * Wa2[j];
        }
        logit[t] = acc;
    }
    float mx = fmaxf(logit[0], logit[1]);
    float w0 = expf(logit[0] - mx), w1 = expf(logit[1] - mx);
    float inv = 1.0f / (w0 + w1);
    w0 *= inv; w1 *= inv;
    for (int k = 0; k < 64; ++k)
        outb[(long)i * 64 + k] = (bf16)(w0 * e0[k] + w1 * e1[k]);
}

// ---------------------------------------------------------------------------
// Tiled bf16 WMMA GEMM:  C[M,Nc] = A[M,Kp] @ Bt[Nc,Kp]^T  (f32 accum)
// 256 threads = 8 waves (4 row x 2 col); each wave: 16 rows x 32 cols
// (two 16x16 WMMA tiles sharing the A fragment).
// Software pipeline: global loads for tile i+1 are issued right after the
// barrier of tile i and only waited on at the LDS store one iteration later,
// so the VMEM latency is hidden behind a full tile of ds_load+WMMA work.
// Out-of-range rows/cols are CLAMPED (not zero-filled): garbage only lands in
// D rows/cols that the epilogue never stores; K is pre-padded with zeros.
// Epilogues: 0 f32 | 1 +bias f32+bf16 | 2 +bias BN ReLU bf16 |
//            3 +bias sigmoid f32 | 4 +bias softplus-clip f32
// ---------------------------------------------------------------------------
template <int EPI>
__global__ __launch_bounds__(256)
void gemm_wmma_bf16(const bf16* __restrict__ A, const bf16* __restrict__ Bt,
                    float* __restrict__ outF, bf16* __restrict__ outB,
                    const float* __restrict__ bias,
                    const float* __restrict__ bn_g, const float* __restrict__ bn_b,
                    const float* __restrict__ bn_rm, const float* __restrict__ bn_rv,
                    int M, int Nc, int Kp) {
    __shared__ bf16 As[2][TM * TK];
    __shared__ bf16 Bs[2][TN * TK];

    const int tid = threadIdx.x;
    const int wave = tid >> 5;
    const int lane = tid & 31;
    const int waveRow = wave >> 1;     // 0..3 -> 16-row slice
    const int waveCol = wave & 1;      // 0..1 -> 32-col slice
    const int half = lane >> 4;        // K-half select per WMMA layout
    const int rc = lane & 15;

    const int rowBase = blockIdx.y * TM;
    const int colBase = blockIdx.x * TN;

    // staging addresses (clamped -> no exec-mask guards in the hot loop)
    const int r = tid >> 2;            // 0..63
    const int c = (tid & 3) * 8;       // 0,8,16,24
    const int rowC = (rowBase + r < M) ? (rowBase + r) : (M - 1);
    const int colC = (colBase + r < Nc) ? (colBase + r) : (Nc - 1);
    const bf16* aPtr = A + (long)rowC * Kp + c;
    const bf16* bPtr = Bt + (long)colC * Kp + c;

    v8f acc0 = {};
    v8f acc1 = {};

    const int nk = Kp / TK;
    v8bf ra = *(const v8bf*)aPtr;      // prologue: issue loads for tile 0
    v8bf rb = *(const v8bf*)bPtr;

    for (int it = 0; it < nk; ++it) {
        const int cur = it & 1;
        // commit staged registers to LDS (loadcnt waited here, one iter late)
        *(v8bf*)&As[cur][r * TK + c] = ra;
        *(v8bf*)&Bs[cur][r * TK + c] = rb;
        __syncthreads();

        // issue next tile's global loads; wait happens next iteration
        if (it + 1 < nk) {
            ra = *(const v8bf*)(aPtr + (long)(it + 1) * TK);
            rb = *(const v8bf*)(bPtr + (long)(it + 1) * TK);
            if (it + 2 < nk) __builtin_prefetch(aPtr + (long)(it + 2) * TK, 0, 1);
        }

        // compute current tile
        union { v16bf v; v8bf h[2]; } fa, fb0, fb1;
        const bf16* ap = &As[cur][(waveRow * 16 + rc) * TK + half * 8];
        fa.h[0] = *(const v8bf*)ap;
        fa.h[1] = *(const v8bf*)(ap + 16);
        const bf16* bp0 = &Bs[cur][(waveCol * 32 + rc) * TK + half * 8];
        fb0.h[0] = *(const v8bf*)bp0;
        fb0.h[1] = *(const v8bf*)(bp0 + 16);
        const bf16* bp1 = bp0 + 16 * TK;
        fb1.h[0] = *(const v8bf*)bp1;
        fb1.h[1] = *(const v8bf*)(bp1 + 16);

        acc0 = __builtin_amdgcn_wmma_f32_16x16x32_bf16(
                   false, fa.v, false, fb0.v, (short)0, acc0, false, false);
        acc1 = __builtin_amdgcn_wmma_f32_16x16x32_bf16(
                   false, fa.v, false, fb1.v, (short)0, acc1, false, false);
    }

    // ---- epilogue ----
    auto apply = [&](float t, int col) -> float {
        if (EPI >= 1) t += bias[col];
        if (EPI == 2) {
            t = (t - bn_rm[col]) * rsqrtf(bn_rv[col] + 1e-5f) * bn_g[col] + bn_b[col];
            t = fmaxf(t, 0.0f);
        } else if (EPI == 3) {
            t = 1.0f / (1.0f + expf(-t));
        } else if (EPI == 4) {
            float sp = (t > 20.0f) ? t : log1pf(expf(t));
            t = fminf(fmaxf(sp, 1e-4f), 1e4f);
        }
        return t;
    };

    const int col0 = colBase + waveCol * 32 + rc;
    const int col1 = col0 + 16;
    #pragma unroll
    for (int v = 0; v < 8; ++v) {
        int row = rowBase + waveRow * 16 + half * 8 + v;
        if (row >= M) continue;
        if (col0 < Nc) {
            float t = apply(acc0[v], col0);
            long o = (long)row * Nc + col0;
            if (EPI == 0 || EPI == 1 || EPI == 3 || EPI == 4) outF[o] = t;
            if (EPI == 1 || EPI == 2) outB[o] = (bf16)t;
        }
        if (col1 < Nc) {
            float t = apply(acc1[v], col1);
            long o = (long)row * Nc + col1;
            if (EPI == 0 || EPI == 1 || EPI == 3 || EPI == 4) outF[o] = t;
            if (EPI == 1 || EPI == 2) outB[o] = (bf16)t;
        }
    }
}

// ---------------------------------------------------------------------------
// Host launcher
// ---------------------------------------------------------------------------
extern "C" void kernel_launch(void* const* d_in, const int* in_sizes, int n_in,
                              void* d_out, int out_size, void* d_ws, size_t ws_size,
                              hipStream_t stream) {
    const int N = 20000, F = 2000, H1 = 256, H2 = 64;
    const int Fp = 2016;               // F padded to multiple of TK
    const int E = in_sizes[1];

    const float* x     = (const float*)d_in[0];
    const int*   s_src = (const int*)d_in[1];
    const int*   s_dst = (const int*)d_in[2];
    const float* s_w   = (const float*)d_in[3];
    const int*   f_src = (const int*)d_in[4];
    const int*   f_dst = (const int*)d_in[5];
    const float* f_w   = (const float*)d_in[6];
    const float* Ws1 = (const float*)d_in[7];  const float* bs1 = (const float*)d_in[8];
    const float* Ws2 = (const float*)d_in[9];  const float* bs2 = (const float*)d_in[10];
    const float* Wf1 = (const float*)d_in[11]; const float* bf1 = (const float*)d_in[12];
    const float* Wf2 = (const float*)d_in[13]; const float* bf2 = (const float*)d_in[14];
    // d_in[15..18] = Wc1,bc1,Wc2,bc2 : results unused in reference, skipped
    const float* Wa1 = (const float*)d_in[19]; const float* ba1 = (const float*)d_in[20];
    const float* Wa2 = (const float*)d_in[21];
    const float* Wm  = (const float*)d_in[22]; const float* bm  = (const float*)d_in[23];
    const float* Wd  = (const float*)d_in[24]; const float* bd  = (const float*)d_in[25];
    const float* bn_g  = (const float*)d_in[26]; const float* bn_b  = (const float*)d_in[27];
    const float* bn_rm = (const float*)d_in[28]; const float* bn_rv = (const float*)d_in[29];
    const float* Wpi = (const float*)d_in[30]; const float* bpi = (const float*)d_in[31];
    const float* Wdp = (const float*)d_in[32]; const float* bdp = (const float*)d_in[33];
    const float* Wmu = (const float*)d_in[34]; const float* bmu = (const float*)d_in[35];

    float* out    = (float*)d_out;
    float* o_emb1 = out;
    float* o_emb2 = out + (long)N * H2;
    float* o_emb  = out + 2L * N * H2;
    float* o_pi   = out + 3L * N * H2;
    float* o_disp = o_pi + (long)N * F;
    float* o_mean = o_disp + (long)N * F;

    // workspace carve (256B aligned)
    char* p = (char*)d_ws;
    auto take = [&](size_t bytes) -> void* {
        void* r = (void*)p;
        p += (bytes + 255) & ~(size_t)255;
        return r;
    };
    bf16* xb    = (bf16*)take((size_t)N * Fp * 2);      // x padded bf16 [N][Fp]
    bf16* Ws1t  = (bf16*)take((size_t)H1 * Fp * 2);     // [256][2016]
    bf16* Wf1t  = (bf16*)take((size_t)H1 * Fp * 2);
    bf16* Ws2t  = (bf16*)take((size_t)H2 * H1 * 2);     // [64][256]
    bf16* Wf2t  = (bf16*)take((size_t)H2 * H1 * 2);
    bf16* Wmt   = (bf16*)take((size_t)H2 * H2 * 2);     // [64][64]
    bf16* Wdt   = (bf16*)take((size_t)H1 * H2 * 2);     // [256][64]
    bf16* Wpit  = (bf16*)take((size_t)F * H1 * 2);      // [2000][256]
    bf16* Wdpt  = (bf16*)take((size_t)F * H1 * 2);
    bf16* Wmut  = (bf16*)take((size_t)F * H1 * 2);
    float* xw   = (float*)take((size_t)N * H1 * 4);
    float* acc1 = (float*)take((size_t)N * H1 * 4);
    bf16*  h1b  = (bf16*)take((size_t)N * H1 * 2);
    float* hw2  = (float*)take((size_t)N * H2 * 4);
    float* acc2 = (float*)take((size_t)N * H2 * 4);
    bf16*  eab  = (bf16*)take((size_t)N * H2 * 2);
    bf16*  embb = (bf16*)take((size_t)N * H2 * 2);
    bf16*  hdb  = (bf16*)take((size_t)N * H1 * 2);

    auto ggrid = [](int M, int Nc) { return dim3((Nc + TN - 1) / TN, (M + TM - 1) / TM); };

    // --- conversions ---
    cvt_pad_bf16<<<dim3((Fp + 255) / 256, N), 256, 0, stream>>>(x, xb, F, Fp);
    cvt_transpose_bf16<<<dim3((Fp + 255) / 256, H1), 256, 0, stream>>>(Ws1, Ws1t, F, H1, Fp);
    cvt_transpose_bf16<<<dim3((Fp + 255) / 256, H1), 256, 0, stream>>>(Wf1, Wf1t, F, H1, Fp);
    cvt_transpose_bf16<<<dim3(1, H2), 256, 0, stream>>>(Ws2, Ws2t, H1, H2, H1);
    cvt_transpose_bf16<<<dim3(1, H2), 256, 0, stream>>>(Wf2, Wf2t, H1, H2, H1);
    cvt_transpose_bf16<<<dim3(1, H2), 256, 0, stream>>>(Wm, Wmt, H2, H2, H2);
    cvt_transpose_bf16<<<dim3(1, H1), 256, 0, stream>>>(Wd, Wdt, H2, H1, H2);
    cvt_transpose_bf16<<<dim3(1, F), 256, 0, stream>>>(Wpi, Wpit, H1, F, H1);
    cvt_transpose_bf16<<<dim3(1, F), 256, 0, stream>>>(Wdp, Wdpt, H1, F, H1);
    cvt_transpose_bf16<<<dim3(1, F), 256, 0, stream>>>(Wmu, Wmut, H1, F, H1);

    // --- S-graph GCN -> emb1 ---
    gemm_wmma_bf16<0><<<ggrid(N, H1), 256, 0, stream>>>(
        xb, Ws1t, xw, nullptr, nullptr, nullptr, nullptr, nullptr, nullptr, N, H1, Fp);
    hipMemsetAsync(acc1, 0, (size_t)N * H1 * 4, stream);
    spmm_edge<<<E, H1, 0, stream>>>(s_src, s_dst, s_w, xw, acc1, H1);
    bias_relu_cvt<<<(unsigned)(((long)N * H1 + 255) / 256), 256, 0, stream>>>(
        acc1, bs1, h1b, (long)N * H1, H1 - 1);
    gemm_wmma_bf16<0><<<ggrid(N, H2), 256, 0, stream>>>(
        h1b, Ws2t, hw2, nullptr, nullptr, nullptr, nullptr, nullptr, nullptr, N, H2, H1);
    hipMemsetAsync(acc2, 0, (size_t)N * H2 * 4, stream);
    spmm_edge<<<E, H2, 0, stream>>>(s_src, s_dst, s_w, hw2, acc2, H2);
    bias_store<<<(unsigned)(((long)N * H2 + 255) / 256), 256, 0, stream>>>(
        acc2, bs2, o_emb1, (long)N * H2, H2 - 1);

    // --- F-graph GCN -> emb2 (reuse scratch) ---
    gemm_wmma_bf16<0><<<ggrid(N, H1), 256, 0, stream>>>(
        xb, Wf1t, xw, nullptr, nullptr, nullptr, nullptr, nullptr, nullptr, N, H1, Fp);
    hipMemsetAsync(acc1, 0, (size_t)N * H1 * 4, stream);
    spmm_edge<<<E, H1, 0, stream>>>(f_src, f_dst, f_w, xw, acc1, H1);
    bias_relu_cvt<<<(unsigned)(((long)N * H1 + 255) / 256), 256, 0, stream>>>(
        acc1, bf1, h1b, (long)N * H1, H1 - 1);
    gemm_wmma_bf16<0><<<ggrid(N, H2), 256, 0, stream>>>(
        h1b, Wf2t, hw2, nullptr, nullptr, nullptr, nullptr, nullptr, nullptr, N, H2, H1);
    hipMemsetAsync(acc2, 0, (size_t)N * H2 * 4, stream);
    spmm_edge<<<E, H2, 0, stream>>>(f_src, f_dst, f_w, hw2, acc2, H2);
    bias_store<<<(unsigned)(((long)N * H2 + 255) / 256), 256, 0, stream>>>(
        acc2, bf2, o_emb2, (long)N * H2, H2 - 1);

    // --- attention fusion ---
    attention_fuse<<<(N + 255) / 256, 256, 0, stream>>>(o_emb1, o_emb2, Wa1, ba1, Wa2, eab, N);

    // --- emb = att@Wm + bm  (f32 to d_out, bf16 for next stage) ---
    gemm_wmma_bf16<1><<<ggrid(N, H2), 256, 0, stream>>>(
        eab, Wmt, o_emb, embb, bm, nullptr, nullptr, nullptr, nullptr, N, H2, H2);

    // --- decoder hidden: relu(BN(emb@Wd + bd)) -> bf16 ---
    gemm_wmma_bf16<2><<<ggrid(N, H1), 256, 0, stream>>>(
        embb, Wdt, nullptr, hdb, bd, bn_g, bn_b, bn_rm, bn_rv, N, H1, H2);

    // --- ZINB heads ---
    gemm_wmma_bf16<3><<<ggrid(N, F), 256, 0, stream>>>(
        hdb, Wpit, o_pi, nullptr, bpi, nullptr, nullptr, nullptr, nullptr, N, F, H1);
    gemm_wmma_bf16<4><<<ggrid(N, F), 256, 0, stream>>>(
        hdb, Wdpt, o_disp, nullptr, bdp, nullptr, nullptr, nullptr, nullptr, N, F, H1);
    gemm_wmma_bf16<3><<<ggrid(N, F), 256, 0, stream>>>(
        hdb, Wmut, o_mean, nullptr, bmu, nullptr, nullptr, nullptr, nullptr, N, F, H1);
}
